// Involution_78958678770311
// MI455X (gfx1250) — matrix operand order repaired and checked by reading, not compiled, not run on hardware
//
#include <hip/hip_runtime.h>
#include <stdint.h>

// ---------- types ----------
typedef __bf16 bf16_t;
typedef __attribute__((ext_vector_type(16))) bf16_t v16bf;
typedef __attribute__((ext_vector_type(8)))  float  v8f;

union BFrag {
    v16bf v;
    bf16_t h[16];
    unsigned short u[16];
    unsigned int   d[8];
};

#define C_   128
#define CR   32
#define O_   72      // G*K*K
#define OP   80      // padded to 5x16 WMMA M-tiles
#define HIN  130
#define H_   128
#define NFRAG_U16 (5 * 4 * 32 * 16)   // mt * kc * lane * elem
#define NFRAG_U32 (5 * 4 * 32 * 8)

// ---------------------------------------------------------------------------
// Prep: fold W_kern@W_reduce -> W_comb (80x128, bf16), pre-swizzled into the
// V_WMMA_F32_16X16X32_BF16 A-fragment layout. Also b_comb (f32).
// A layout (ISA 7.12.2, 16-bit A 16x32): lane L holds row M=L&15,
// K = (L>>4)*8 + e + (e>=8 ? 8 : 0) for element e of the 16-elem fragment.
// ---------------------------------------------------------------------------
__global__ void involution_prep(const float* __restrict__ Wr,   // (32,128)
                                const float* __restrict__ br,   // (32)
                                const float* __restrict__ Wk,   // (72,32)
                                const float* __restrict__ bk,   // (72)
                                bf16_t* __restrict__ wsA,
                                float* __restrict__ wsB) {
    const int tid = threadIdx.x;
    for (int t = tid; t < NFRAG_U16; t += 256) {
        int e    = t & 15;
        int lane = (t >> 4) & 31;
        int kc   = (t >> 9) & 3;
        int mt   = t >> 11;
        int m    = lane & 15;
        int k    = (lane >> 4) * 8 + e + (e >= 8 ? 8 : 0);
        int o    = mt * 16 + m;
        int c    = kc * 32 + k;
        float v = 0.f;
        if (o < O_) {
            #pragma unroll
            for (int r = 0; r < CR; ++r) v += Wk[o * CR + r] * Wr[r * C_ + c];
        }
        wsA[t] = (bf16_t)v;                       // native v_cvt, RNE
    }
    if (tid < OP) {
        float v = 0.f;
        if (tid < O_) {
            #pragma unroll
            for (int r = 0; r < CR; ++r) v += Wk[tid * CR + r] * br[r];
            v += bk[tid];
        }
        wsB[tid] = v;
    }
}

// ---------------------------------------------------------------------------
// Fused main kernel: one workgroup per (batch, row). 8 waves; each wave owns
// 16 pixels: builds bf16 B-fragments from the center channels, runs 20
// v_wmma_f32_16x16x32_bf16 to generate the 72 per-pixel kernel values, parks
// them in LDS, then applies the 3x3 per-group filter in fp32.
// ---------------------------------------------------------------------------
__global__ void __launch_bounds__(256)
involution_main(const float* __restrict__ x,
                const unsigned short* __restrict__ wsA,
                const float* __restrict__ wsB,
                float* __restrict__ out) {
    __shared__ unsigned int sA[NFRAG_U32];     // 20480 B
    __shared__ float sBias[OP];                //   320 B
    __shared__ float sKern[8][OP][17];         // 43520 B (pad 17: no bank conflicts)

    const int tid    = threadIdx.x;
    const int b      = blockIdx.x >> 7;
    const int py     = blockIdx.x & 127;
    const int wv     = tid >> 5;
    const int lane   = tid & 31;
    const int half   = lane >> 4;              // 0/1 : WMMA half-wave
    const int pxl    = lane & 15;              // N column / pixel in tile
    const int tile_x = wv * 16;
    const int px     = tile_x + pxl;

    // Stage A-fragments + bias once per block
    for (int i = tid; i < NFRAG_U32; i += 256)
        sA[i] = ((const unsigned int*)wsA)[i];
    if (tid < OP) sBias[tid] = wsB[tid];
    __syncthreads();

    const float* xb = x + (size_t)b * C_ * HIN * HIN;

    // B-fragments: xc[:,py,px] -> bf16. B layout (32x16 16-bit): lane L holds
    // column N=L&15, K = (L>>4)*8 + e + (e>=8 ? 8 : 0). Native converts (v_cvt_pk_bf16_f32).
    BFrag bfrag[4];
    #pragma unroll
    for (int kc = 0; kc < 4; ++kc) {
        #pragma unroll
        for (int e = 0; e < 16; ++e) {
            int ch  = kc * 32 + half * 8 + e + (e >= 8 ? 8 : 0);
            float v = xb[(size_t)ch * (HIN * HIN) + (size_t)(py + 1) * HIN + (px + 1)];
            bfrag[kc].h[e] = (bf16_t)v;
        }
    }

    // kern = W_comb @ xc + b_comb : 5 M-tiles x 4 K-chunks = 20 WMMAs
    #pragma unroll
    for (int mt = 0; mt < 5; ++mt) {
        v8f acc;
        #pragma unroll
        for (int i = 0; i < 8; ++i)
            acc[i] = sBias[mt * 16 + half * 8 + i];     // C/D: M = i + 8*half
        #pragma unroll
        for (int kc = 0; kc < 4; ++kc) {
            BFrag a;
            const unsigned int* src = &sA[(size_t)((mt * 4 + kc) * 32 + lane) * 8];
            #pragma unroll
            for (int d = 0; d < 8; ++d) a.d[d] = src[d];
            acc = __builtin_amdgcn_wmma_f32_16x16x32_bf16(
                      false, a.v, false, bfrag[kc].v, (short)0, acc, false, false);
        }
        #pragma unroll
        for (int i = 0; i < 8; ++i)
            sKern[wv][mt * 16 + half * 8 + i][pxl] = acc[i];
    }
    // Same-wave LDS RAW across lanes; compiler orders via DScnt waits.

    // Apply: out[b,ch,py,px] = sum_k kern[g*9+k] * x[b,ch,py+di,px+dj]
    const float* xrow = xb + (size_t)py * HIN + (tile_x + pxl);
    for (int g = 0; g < 8; ++g) {
        float kv[9];
        #pragma unroll
        for (int k = 0; k < 9; ++k) kv[k] = sKern[wv][g * 9 + k][pxl];
        #pragma unroll
        for (int cc = 0; cc < 8; ++cc) {
            int ch = g * 16 + cc * 2 + half;     // lanes 0-15: even, 16-31: odd
            const float* xp = xrow + (size_t)ch * (HIN * HIN);
            float sum = 0.f;
            #pragma unroll
            for (int di = 0; di < 3; ++di)
                #pragma unroll
                for (int dj = 0; dj < 3; ++dj)
                    sum += kv[di * 3 + dj] * xp[di * HIN + dj];
            // Streamed output: non-temporal store, keep L2 for x re-reads.
            __builtin_nontemporal_store(
                sum, &out[(((size_t)b * C_ + ch) * H_ + py) * H_ + (tile_x + pxl)]);
        }
    }
}

// ---------------------------------------------------------------------------
extern "C" void kernel_launch(void* const* d_in, const int* in_sizes, int n_in,
                              void* d_out, int out_size, void* d_ws, size_t ws_size,
                              hipStream_t stream) {
    const float* x  = (const float*)d_in[0];
    const float* Wr = (const float*)d_in[1];
    const float* br = (const float*)d_in[2];
    const float* Wk = (const float*)d_in[3];
    const float* bk = (const float*)d_in[4];

    bf16_t* wsA = (bf16_t*)d_ws;                                 // 20480 B
    float*  wsB = (float*)((char*)d_ws + NFRAG_U16 * 2);         //   320 B
    float*  out = (float*)d_out;

    hipLaunchKernelGGL(involution_prep, dim3(1), dim3(256), 0, stream,
                       Wr, br, Wk, bk, wsA, wsB);
    hipLaunchKernelGGL(involution_main, dim3(8 * H_), dim3(256), 0, stream,
                       x, (const unsigned short*)wsA, wsB, out);
}
